// SpikingReservoir_29042568855714
// MI455X (gfx1250) — matrix operand order repaired and checked by reading, not compiled, not run on hardware
//
#include <hip/hip_runtime.h>

// Problem constants (match reference)
#define B_ 64
#define T_ 500
#define N_ 2048
#define BETA_RES 0.9f
#define BETA_IN  0.8f
#define THRESH   1.0f

typedef __bf16 bf16_t;
typedef __attribute__((ext_vector_type(16))) __bf16 v16bf;
typedef __attribute__((ext_vector_type(8)))  float  v8f;
typedef __attribute__((ext_vector_type(4)))  unsigned int u32x4;

union V16BF { v16bf v; u32x4 q[2]; };

// ---------------------------------------------------------------------------
// 1) Input-LIF scalar scan: imem_t = BETA_IN*imem_{t-1} + x[b,t]*w_in.
//    Independent of the reservoir -> precompute all (t,b) once.
// ---------------------------------------------------------------------------
__global__ void imem_precompute(const float* __restrict__ x,
                                const float* __restrict__ w_in,
                                float* __restrict__ imem_all) {
  int b = threadIdx.x;
  if (b >= B_) return;
  float w = w_in[0];
  float imem = 0.0f;
  for (int t = 0; t < T_; ++t) {
    imem = BETA_IN * imem + x[b * T_ + t] * w;   // x is (B,T,1)
    imem_all[t * B_ + b] = imem;
  }
}

// ---------------------------------------------------------------------------
// 2) W_rec f32 -> bf16 (row-major, W[n][k]); 8 MB, stays L2-resident.
// ---------------------------------------------------------------------------
__global__ void convert_w(const float* __restrict__ W, bf16_t* __restrict__ Wbf) {
  int i = blockIdx.x * blockDim.x + threadIdx.x;
  if (i < N_ * N_) Wbf[i] = (bf16_t)W[i];
}

// ---------------------------------------------------------------------------
// 3) Zero the reservoir state (mem=0, spk=0) + spike counter.
// ---------------------------------------------------------------------------
__global__ void init_state(float* __restrict__ mem, bf16_t* __restrict__ spk0,
                           float* __restrict__ counter) {
  int i = blockIdx.x * blockDim.x + threadIdx.x;
  if (i < B_ * N_) { mem[i] = 0.0f; spk0[i] = (bf16_t)0.0f; }
  if (i == 0) *counter = 0.0f;
}

// ---------------------------------------------------------------------------
// 4) One reservoir time step.
//    C[b,n] = sum_k spk[b,k] * W[n,k]   via v_wmma_f32_16x16x32_bf16.
//    Grid: 128 blocks (N-tiles of 16 cols), 4 waves/block (M-tiles of 16 rows).
//    A (16x32 bf16), lane l: row m = l&15, K halves at (l>>4)*8 and +16
//      -> two b128 loads from row-major spk_bf.
//    B (32x16 bf16), lane l: col n = l&15, K halves at (l>>4)*8 and +16
//      -> two b128 loads from row-major W_bf (B[k,n] = W[n,k]).
//    C/D layout: lane = col (l&15), VGPR r = row r + 8*(l>>4).
//    Record writes use non-temporal stores (write-once streaming data) so the
//    0.52 GB of spike/mem records doesn't evict the L2-resident W/state.
// ---------------------------------------------------------------------------
__global__ __launch_bounds__(128) void reservoir_step(
    int t,
    const bf16_t* __restrict__ Wbf,
    const bf16_t* __restrict__ spkSrc,
    bf16_t*       __restrict__ spkDst,
    float*        __restrict__ mem,
    const float*  __restrict__ imem_all,
    const float*  __restrict__ w_res,
    float*        __restrict__ spk_rec,   // d_out + 1
    float*        __restrict__ mem_rec,   // d_out + 1 + T*B*N
    float*        __restrict__ counter) {
  const int lane  = threadIdx.x & 31;
  const int mtile = threadIdx.x >> 5;   // 0..3  (M tile = batch rows)
  const int ntile = blockIdx.x;         // 0..127 (N tile = neuron cols)
  const int hi    = lane >> 4;          // 0/1 : lane half
  const int l15   = lane & 15;

  // A-tile base: row m, k-offset hi*8 (loads cover k..k+7 and k+16..k+23)
  const int am = mtile * 16 + l15;
  const bf16_t* aRow = spkSrc + (size_t)am * N_ + hi * 8;
  // B-tile base: col n, k-offset hi*8 (loads cover k..k+7 and k+16..k+23)
  const int n = ntile * 16 + l15;
  const bf16_t* bRow = Wbf + (size_t)n * N_ + hi * 8;

  v8f acc = {};
#pragma unroll 4
  for (int kt = 0; kt < N_ / 32; ++kt) {
    V16BF a, b;
    a.q[0] = *(const u32x4*)(aRow + kt * 32);
    a.q[1] = *(const u32x4*)(aRow + kt * 32 + 16);
    b.q[0] = *(const u32x4*)(bRow + kt * 32);
    b.q[1] = *(const u32x4*)(bRow + kt * 32 + 16);
    acc = __builtin_amdgcn_wmma_f32_16x16x32_bf16(
        /*neg_a=*/false, a.v, /*neg_b=*/false, b.v,
        /*c_mod=*/(short)0, acc, /*reuse_a=*/false, /*reuse_b=*/false);
  }

  // ---- LIF update in the C/D register layout ----
  const int m0 = mtile * 16 + hi * 8;
  const float wres = w_res[n];
  const float* imemT = imem_all + t * B_;
  const size_t recOff = (size_t)t * B_ * N_;
  float cnt = 0.0f;
#pragma unroll
  for (int r = 0; r < 8; ++r) {
    const int m = m0 + r;                       // batch index
    const int idx = m * N_ + n;
    float mv = mem[idx];
    float sp_prev = (mv > THRESH) ? 1.0f : 0.0f;     // zero-reset gate
    float nm = BETA_RES * mv * (1.0f - sp_prev) + imemT[m] * wres + acc[r];
    float sp = (nm > THRESH) ? 1.0f : 0.0f;
    mem[idx] = nm;                              // re-read next step: keep cached
    spkDst[idx] = (bf16_t)sp;                   // A-operand for step t+1: cached
    // Write-once streaming records -> non-temporal (TH=NT) stores
    __builtin_nontemporal_store(nm, mem_rec + recOff + idx);
    __builtin_nontemporal_store(sp, spk_rec + recOff + idx);
    cnt += sp;
  }
  // wave32 reduction of spike count, one atomic per wave
  for (int o = 16; o > 0; o >>= 1) cnt += __shfl_xor(cnt, o, 32);
  if (lane == 0) atomicAdd(counter, cnt);
}

// ---------------------------------------------------------------------------
// 5) avg_firing_rate = total_spikes / (T*B*N)
// ---------------------------------------------------------------------------
__global__ void finalize(const float* __restrict__ counter, float* __restrict__ out0) {
  if (threadIdx.x == 0 && blockIdx.x == 0)
    out0[0] = counter[0] / (float)((long long)T_ * B_ * N_);
}

// ---------------------------------------------------------------------------
extern "C" void kernel_launch(void* const* d_in, const int* in_sizes, int n_in,
                              void* d_out, int out_size, void* d_ws, size_t ws_size,
                              hipStream_t stream) {
  const float* x     = (const float*)d_in[0];   // (B,T,1)
  const float* w_in  = (const float*)d_in[1];   // (1,1)
  const float* w_res = (const float*)d_in[2];   // (N,1)
  const float* W_rec = (const float*)d_in[3];   // (N,N)
  float* out = (float*)d_out;                   // [1 + T*B*N + T*B*N]

  // Workspace carve-up (~9.6 MB total)
  char* ws = (char*)d_ws;
  size_t o = 0;
  bf16_t* Wbf      = (bf16_t*)(ws + o); o += (size_t)N_ * N_ * sizeof(bf16_t);
  float*  mem      = (float*)(ws + o);  o += (size_t)B_ * N_ * sizeof(float);
  bf16_t* spkA     = (bf16_t*)(ws + o); o += (size_t)B_ * N_ * sizeof(bf16_t);
  bf16_t* spkB     = (bf16_t*)(ws + o); o += (size_t)B_ * N_ * sizeof(bf16_t);
  float*  imem_all = (float*)(ws + o);  o += (size_t)T_ * B_ * sizeof(float);
  float*  counter  = (float*)(ws + o);  o += sizeof(float);

  imem_precompute<<<1, 64, 0, stream>>>(x, w_in, imem_all);
  convert_w<<<(N_ * N_ + 255) / 256, 256, 0, stream>>>(W_rec, Wbf);
  init_state<<<(B_ * N_ + 255) / 256, 256, 0, stream>>>(mem, spkA, counter);

  float* spk_rec = out + 1;
  float* mem_rec = out + 1 + (size_t)T_ * B_ * N_;
  for (int t = 0; t < T_; ++t) {
    const bf16_t* src = (t & 1) ? spkB : spkA;
    bf16_t*       dst = (t & 1) ? spkA : spkB;
    reservoir_step<<<dim3(N_ / 16), dim3(128), 0, stream>>>(
        t, Wbf, src, dst, mem, imem_all, w_res, spk_rec, mem_rec, counter);
  }
  finalize<<<1, 1, 0, stream>>>(counter, out);
}
